// Torch_RNN_5609227288755
// MI455X (gfx1250) — compile-verified, matrix operand order checked by reading, not run on hardware
//
#include <hip/hip_runtime.h>

typedef _Float16 v16h __attribute__((ext_vector_type(16)));
typedef _Float16 v8h  __attribute__((ext_vector_type(8)));
typedef _Float16 v4h  __attribute__((ext_vector_type(4)));
typedef _Float16 v2h  __attribute__((ext_vector_type(2)));
typedef __fp16   v2hf __attribute__((ext_vector_type(2)));  // cvt_pkrtz result type
typedef float    v8f  __attribute__((ext_vector_type(8)));
typedef float    v4f  __attribute__((ext_vector_type(4)));

#define INDIM   28
#define HID     128
#define OUTDIM  10
#define TSTEPS  28
#define XCOLS   784
#define ROWS_PER_BLOCK 256
#define THREADS 256

// ---- LDS layout (units of _Float16 unless noted) ----
// steady state:
//   sW12  : 8 tiles * 512    B-fragments of [W1@W2a ; c2] (K 28(+bias row)->32, N=128)
//   sW2b  : 32 tiles * 512   B-fragments of W2[128:] (128x128)
//   sW3   : 4 tiles * 512    B-fragments of W3 (128x10, N padded to 16)
//   sX    : 256 rows * 32    x tile in A-fragment element order
//                            (cols permuted [0..7,16..23,8..15,24..31], col28=1.0 bias)
//   sStage: 8 waves * 2 tiles * 2048   per-wave D->A transpose staging for h
//   sB3   : 16 f32
// setup temps (fp32, aliased over sW2b.., barrier-separated):
//   sW2aF : 128*128 f32,  sW1F : 28*128 f32
#define H_W12   (8*512)
#define OFF_B   H_W12                 // 4096 halfs
#define H_W2B   (32*512)
#define OFF_W3  (OFF_B + H_W2B)       // 20480
#define H_W3    (4*512)
#define OFF_X   (OFF_W3 + H_W3)       // 22528
#define H_X     (256*32)              // 8192
#define OFF_STG (OFF_X + H_X)         // 30720
#define H_STG   (8*4096)              // ends at 63488 halfs
#define OFF_B3_BYTES   (63488*2)      // 126976
#define OFF_W2AF_BYTES (OFF_B*2)      // 8192
#define OFF_W1F_BYTES  (OFF_W2AF_BYTES + 128*128*4)   // 73728
#define SMEM_BYTES ((size_t)(OFF_B3_BYTES + 64))      // 127040 (> temps end 88064)

static __device__ __forceinline__ int koff_of(int i, int lane) {
  // 16-bit A/B fragment K offset within a 32-wide K tile (ISA 7.12.2)
  return i + ((i >= 8) ? 8 : 0) + ((lane >= 16) ? 8 : 0);
}

static __device__ __forceinline__ v2h relu_pk(v2hf p) {
  v2hf r;
  asm("v_pk_max_num_f16 %0, %1, 0" : "=v"(r) : "v"(p));
  return __builtin_bit_cast(v2h, r);
}

__global__ __launch_bounds__(THREADS)
void rnn_wmma_kernel(const float* __restrict__ x,
                     const float* __restrict__ W1,
                     const float* __restrict__ b1,
                     const float* __restrict__ W2,
                     const float* __restrict__ b2,
                     const float* __restrict__ W3,
                     const float* __restrict__ b3,
                     float* __restrict__ out) {
  extern __shared__ __align__(128) _Float16 smem[];
  _Float16* sW12   = smem;
  _Float16* sW2b   = smem + OFF_B;
  _Float16* sW3    = smem + OFF_W3;
  _Float16* sX     = smem + OFF_X;
  _Float16* sStage = smem + OFF_STG;
  float*    sB3    = (float*)((char*)smem + OFF_B3_BYTES);
  float*    sW2aF  = (float*)((char*)smem + OFF_W2AF_BYTES);
  float*    sW1F   = (float*)((char*)smem + OFF_W1F_BYTES);

  const int tid  = threadIdx.x;
  const int wave = tid >> 5;
  const int lane = tid & 31;
  const int ln15 = lane & 15;
  const int hi   = lane >> 4;

  // ================= SETUP =================
  // Phase 1: stage fp32 W1 / W2a into LDS (coalesced float4), zero sW12
  for (int p = tid; p < (HID*HID)/4; p += THREADS)
    ((v4f*)sW2aF)[p] = ((const v4f*)W2)[p];
  for (int p = tid; p < (INDIM*HID)/4; p += THREADS)
    ((v4f*)sW1F)[p] = ((const v4f*)W1)[p];
  for (int p = tid; p < H_W12; p += THREADS)
    sW12[p] = (_Float16)0.0f;
  __syncthreads();

  // Phase 2: fp32 fold  W12 = W1@W2a,  c2 = b1@W2a + b2; scatter into B-fragments
  {
    const int n  = tid & 127;
    const int kb = (tid >> 7) * 14;              // rows kb..kb+13 of 28
    float acc[14];
#pragma unroll
    for (int kk = 0; kk < 14; ++kk) acc[kk] = 0.0f;
    for (int j = 0; j < HID; ++j) {
      float wv = sW2aF[j*HID + n];
#pragma unroll
      for (int kk = 0; kk < 14; ++kk)
        acc[kk] += sW1F[(kb + kk)*HID + j] * wv;
    }
    float c2 = 0.0f;
    if (tid < HID) {
      c2 = b2[tid];
      for (int j = 0; j < HID; ++j) c2 += b1[j] * sW2aF[j*HID + tid];
    }
#pragma unroll
    for (int kk = 0; kk < 14; ++kk) {            // B-fragment scatter of W12
      int k  = kb + kk;
      int i  = k - ((k >= 8) ? 8 : 0) - ((k >= 24) ? 8 : 0);
      int lp = ((k >> 3) & 1) << 4;
      sW12[(n >> 4)*512 + ((n & 15) + lp)*16 + i] = (_Float16)acc[kk];
    }
    if (tid < HID) {                             // c2 rides K row 28 (i=12, hi lanes)
      sW12[(tid >> 4)*512 + ((tid & 15) + 16)*16 + 12] = (_Float16)c2;
    }
  }
  __syncthreads();                               // temp fp32 regions now dead

  // Phase 3: B-fragments of W2b and W3 (from global), x pad columns, b3
  for (int p = tid; p < H_W2B; p += THREADS) {
    int tile = p >> 9, w = p & 511, ln = w >> 4, i = w & 15;
    int kt = tile >> 3, nt = tile & 7;
    int k = kt*32 + koff_of(i, ln);
    int n = nt*16 + (ln & 15);
    sW2b[p] = (_Float16)W2[(k + HID)*HID + n];
  }
  for (int p = tid; p < H_W3; p += THREADS) {
    int w = p & 511, ln = w >> 4, i = w & 15, kt = p >> 9;
    int k = kt*32 + koff_of(i, ln);
    int n = ln & 15;
    float v = (n < OUTDIM) ? W3[k*OUTDIM + n] : 0.0f;
    sW3[p] = (_Float16)v;
  }
  for (int p = tid; p < ROWS_PER_BLOCK; p += THREADS) {  // bias/pad cols, once
    v4h pad; pad[0] = (_Float16)1.0f; pad[1] = (_Float16)0.0f;
    pad[2] = (_Float16)0.0f; pad[3] = (_Float16)0.0f;
    *(v4h*)&sX[p*32 + 28] = pad;                 // perm(28..31) == 28..31
  }
  if (tid < 16) sB3[tid] = (tid < OUTDIM) ? b3[tid] : 0.0f;

  // ================= MAIN RECURRENCE =================
  // Each wave owns TWO 16-row batch tiles: every B-fragment load feeds 2 WMMAs.
  const long rowBase   = (long)blockIdx.x * ROWS_PER_BLOCK;
  const int  waveRow   = wave * 32;
  const int  stageBase = wave * 4096;

  v16h hA[2][4];                                 // h state as A fragments (h0 = 0)
#pragma unroll
  for (int u = 0; u < 2; ++u)
#pragma unroll
    for (int kt = 0; kt < 4; ++kt)
#pragma unroll
      for (int j = 0; j < 16; ++j) hA[u][kt][j] = (_Float16)0.0f;

  v8f zero;
#pragma unroll
  for (int r = 0; r < 8; ++r) zero[r] = 0.0f;

  for (int t = 0; t < TSTEPS; ++t) {
    __syncthreads();  // prior step done reading sX (covers setup on t==0)
    // ---- stage x[:, t, :]: 7 aligned float4 per row, stored in fragment order ----
    const float* xt = x + rowBase * XCOLS + t * INDIM;
#pragma unroll
    for (int p = tid; p < ROWS_PER_BLOCK*7; p += THREADS) {
      int row = p / 7, q = p % 7;
      int c   = q * 4;
      int pos = c + ((c < 8) ? 0 : (c < 16) ? 8 : (c < 24) ? -8 : 0);
      v4f f = *(const v4f*)(xt + (long)row * XCOLS + c);
      v2h l2 = __builtin_bit_cast(v2h, __builtin_amdgcn_cvt_pkrtz(f[0], f[1]));
      v2h h2 = __builtin_bit_cast(v2h, __builtin_amdgcn_cvt_pkrtz(f[2], f[3]));
      v4h h4; h4[0] = l2[0]; h4[1] = l2[1]; h4[2] = h2[0]; h4[3] = h2[1];
      *(v4h*)&sX[row*32 + pos] = h4;
    }
    __syncthreads();
    if (t + 1 < TSTEPS)                          // warm caches for next slice
      __builtin_prefetch(xt + INDIM + (long)tid * XCOLS, 0, 0);

    // ---- A fragments of x' tiles: one contiguous 32B v16h load per lane ----
    v16h ax[2];
#pragma unroll
    for (int u = 0; u < 2; ++u)
      ax[u] = *(const v16h*)&sX[(waveRow + u*16 + ln15)*32 + hi*16];

    // ---- per N-tile: acc = x'@W12' + h@W2b, then relu+stage immediately ----
#pragma unroll
    for (int nt = 0; nt < 8; ++nt) {
      v16h bw12 = *(const v16h*)&sW12[nt*512 + lane*16];
      v8f a0 = __builtin_amdgcn_wmma_f32_16x16x32_f16(
          false, ax[0], false, bw12, (short)0, zero, false, false);
      v8f a1 = __builtin_amdgcn_wmma_f32_16x16x32_f16(
          false, ax[1], false, bw12, (short)0, zero, false, false);
#pragma unroll
      for (int kt = 0; kt < 4; ++kt) {
        v16h bb = *(const v16h*)&sW2b[(kt*8 + nt)*512 + lane*16];
        a0 = __builtin_amdgcn_wmma_f32_16x16x32_f16(
            false, hA[0][kt], false, bb, (short)0, a0, false, false);
        a1 = __builtin_amdgcn_wmma_f32_16x16x32_f16(
            false, hA[1][kt], false, bb, (short)0, a1, false, false);
      }
      // pack to f16 (RTZ) then packed ReLU: 1 VALU per value total
      int kk = ((nt & 1) << 4) | ln15;
      int i  = kk - ((kk >= 8) ? 8 : 0) - ((kk >= 24) ? 8 : 0);
      int lp = ((kk >> 3) & 1) << 4;
      _Float16* dst0 = &sStage[stageBase + (nt >> 1)*512 + (lp + (hi << 3))*16 + i];
      _Float16* dst1 = dst0 + 2048;
#pragma unroll
      for (int r = 0; r < 8; r += 2) {
        v2h q0 = relu_pk(__builtin_amdgcn_cvt_pkrtz(a0[r], a0[r+1]));
        dst0[r*16] = q0[0]; dst0[(r+1)*16] = q0[1];
        v2h q1 = relu_pk(__builtin_amdgcn_cvt_pkrtz(a1[r], a1[r+1]));
        dst1[r*16] = q1[0]; dst1[(r+1)*16] = q1[1];
      }
    }
    // reload h state as A fragments
#pragma unroll
    for (int u = 0; u < 2; ++u)
#pragma unroll
      for (int kt = 0; kt < 4; ++kt)
        hA[u][kt] = *(const v16h*)&sStage[stageBase + u*2048 + kt*512 + lane*16];
  }

  // ---- out = h @ W3 + b3 ----
#pragma unroll
  for (int u = 0; u < 2; ++u) {
    v8f ao;
    float bv = sB3[ln15];
#pragma unroll
    for (int r = 0; r < 8; ++r) ao[r] = bv;
#pragma unroll
    for (int kt = 0; kt < 4; ++kt) {
      v16h bw = *(const v16h*)&sW3[kt*512 + lane*16];
      ao = __builtin_amdgcn_wmma_f32_16x16x32_f16(
          false, hA[u][kt], false, bw, (short)0, ao, false, false);
    }
    if (ln15 < OUTDIM) {
#pragma unroll
      for (int r = 0; r < 8; ++r) {
        int m = r + (hi << 3);
        out[(rowBase + waveRow + u*16 + m) * OUTDIM + ln15] = ao[r];
      }
    }
  }
}

extern "C" void kernel_launch(void* const* d_in, const int* in_sizes, int n_in,
                              void* d_out, int out_size, void* d_ws, size_t ws_size,
                              hipStream_t stream) {
  const float* x  = (const float*)d_in[0];
  const float* W1 = (const float*)d_in[1];
  const float* b1 = (const float*)d_in[2];
  const float* W2 = (const float*)d_in[3];
  const float* b2 = (const float*)d_in[4];
  const float* W3 = (const float*)d_in[5];
  const float* b3 = (const float*)d_in[6];
  float* out = (float*)d_out;

  const int B = in_sizes[0] / XCOLS;             // 65536
  dim3 grid(B / ROWS_PER_BLOCK);                 // 256 workgroups
  dim3 block(THREADS);                           // 8 wave32 per workgroup
  rnn_wmma_kernel<<<grid, block, SMEM_BYTES, stream>>>(x, W1, b1, W2, b2, W3, b3, out);
}
